// Head_enc_81063212745013
// MI455X (gfx1250) — compile-verified, matrix operand order checked by reading, not run on hardware
//
#include <hip/hip_runtime.h>

// ---------------------------------------------------------------------------
// Attention head forward (B=8, T=2048, C=768, H=64), fp32 reference.
// bf16 WMMA (v_wmma_f32_16x16x32_bf16, fp32 accumulate) + flash-attention
// fusion (no 134MB score matrix in HBM) + async global->LDS staging with
// double buffering (GLOBAL_LOAD_ASYNC_TO_LDS_B128 / s_wait_asynccnt).
// ---------------------------------------------------------------------------

typedef __attribute__((ext_vector_type(16))) __bf16 v16bf;
typedef __attribute__((ext_vector_type(8)))  __bf16 v8bf;
typedef __attribute__((ext_vector_type(8)))  float  v8f;
typedef __attribute__((ext_vector_type(4)))  int    v4i;

#define AS_GLOBAL __attribute__((address_space(1)))
#define AS_LDS    __attribute__((address_space(3)))

constexpr int BATCH = 8;
constexpr int TLEN  = 2048;
constexpr int CDIM  = 768;
constexpr int HDIM  = 64;
constexpr int BT    = BATCH * TLEN;
constexpr float SM_SCALE = 0.03608439182435161f; // 768^-0.5 (reference scales by C^-0.5)

static __device__ __forceinline__ v16bf cat16(v8bf lo, v8bf hi) {
  v16bf r;
#pragma unroll
  for (int i = 0; i < 8; ++i) { r[i] = lo[i]; r[i + 8] = hi[i]; }
  return r;
}

static __device__ __forceinline__ v8f wmma_bf16(v16bf a, v16bf b, v8f c) {
  // D = A(16x32 bf16) * B(32x16 bf16) + C(16x16 f32)
  return __builtin_amdgcn_wmma_f32_16x16x32_bf16(
      /*neg_a=*/false, a, /*neg_b=*/false, b,
      /*c_mod=*/(short)0, c, /*reuse_a=*/false, /*reuse_b=*/false);
}

static __device__ __forceinline__ void async_copy_16B(const __bf16* gsrc, __bf16* ldst) {
#if __has_builtin(__builtin_amdgcn_global_load_async_to_lds_b128)
  __builtin_amdgcn_global_load_async_to_lds_b128(
      (AS_GLOBAL v4i*)gsrc, (AS_LDS v4i*)ldst, 0, 0);
#else
  *(v8bf*)ldst = *(const v8bf*)gsrc;   // fallback: load->store through VGPRs
#endif
}

static __device__ __forceinline__ void wait_async0() {
#if __has_builtin(__builtin_amdgcn_s_wait_asynccnt)
  __builtin_amdgcn_s_wait_asynccnt(0);
#else
  asm volatile("s_wait_asynccnt 0" ::: "memory");
#endif
}

// ---------------------------------------------------------------------------
// Kernel 1: qkv projection.  [BT x C] fp32  @  [C x H] fp32  ->  [BT x H] bf16
// grid = (BT/64, 3); block = 128 (4 waves, 16 rows each).
// ---------------------------------------------------------------------------
__global__ __launch_bounds__(128)
void proj_kernel(const float* __restrict__ x,
                 const float* __restrict__ Wk,
                 const float* __restrict__ Wq,
                 const float* __restrict__ Wv,
                 __bf16* __restrict__ kbuf,
                 __bf16* __restrict__ qbuf,
                 __bf16* __restrict__ vbuf) {
  // W chunk staged transposed: wTc[h][k_local] so B-fragments read contiguous K.
  __shared__ __attribute__((aligned(16))) __bf16 wTc[HDIM][32];

  const int tid  = threadIdx.x;
  const int wv   = tid >> 5;
  const int lane = tid & 31;
  const int m    = lane & 15;   // tile row (A) / tile col (B,D)
  const int g    = lane >> 4;   // lane half selects K sub-range

  const float* W   = (blockIdx.y == 0) ? Wk : (blockIdx.y == 1) ? Wq : Wv;
  __bf16*      out = (blockIdx.y == 0) ? kbuf : (blockIdx.y == 1) ? qbuf : vbuf;

  const int rowBase = blockIdx.x * 64 + wv * 16;
  const float* xr = x + (size_t)(rowBase + m) * CDIM;

  v8f acc[4] = {{}, {}, {}, {}};

  for (int kc = 0; kc < CDIM / 32; ++kc) {
    const int kb0 = kc * 32;

    // --- stage W[kb0:kb0+32][0:64] transposed into LDS as bf16 ---
    {
      const int klocal = tid >> 2;          // 0..31
      const int hb     = (tid & 3) * 16;    // 0,16,32,48
      const float4* w4 = (const float4*)(W + (size_t)(kb0 + klocal) * HDIM + hb);
      float4 a0 = w4[0], a1 = w4[1], a2 = w4[2], a3 = w4[3];
      const float wf[16] = {a0.x, a0.y, a0.z, a0.w, a1.x, a1.y, a1.z, a1.w,
                            a2.x, a2.y, a2.z, a2.w, a3.x, a3.y, a3.z, a3.w};
#pragma unroll
      for (int i = 0; i < 16; ++i) wTc[hb + i][klocal] = (__bf16)wf[i];
    }
    __syncthreads();

    // --- A fragment: x rows, fp32 -> bf16 on the fly ---
    const float4* x4a = (const float4*)(xr + kb0 + 8 * g);
    const float4* x4b = (const float4*)(xr + kb0 + 16 + 8 * g);
    float4 f0 = x4a[0], f1 = x4a[1], f2 = x4b[0], f3 = x4b[1];
    v16bf a;
    a[0] = (__bf16)f0.x; a[1] = (__bf16)f0.y; a[2] = (__bf16)f0.z; a[3] = (__bf16)f0.w;
    a[4] = (__bf16)f1.x; a[5] = (__bf16)f1.y; a[6] = (__bf16)f1.z; a[7] = (__bf16)f1.w;
    a[8] = (__bf16)f2.x; a[9] = (__bf16)f2.y; a[10] = (__bf16)f2.z; a[11] = (__bf16)f2.w;
    a[12] = (__bf16)f3.x; a[13] = (__bf16)f3.y; a[14] = (__bf16)f3.z; a[15] = (__bf16)f3.w;

#pragma unroll
    for (int h = 0; h < 4; ++h) {
      const __bf16* wr = &wTc[16 * h + m][0];
      v16bf bfrag = cat16(*(const v8bf*)(wr + 8 * g),
                          *(const v8bf*)(wr + 16 + 8 * g));
      acc[h] = wmma_bf16(a, bfrag, acc[h]);
    }
    __syncthreads();
  }

  // D layout: lane col n=m, VGPR r -> row M = r + 8*g
#pragma unroll
  for (int r = 0; r < 8; ++r) {
    __bf16* orow = out + (size_t)(rowBase + r + 8 * g) * HDIM;
#pragma unroll
    for (int h = 0; h < 4; ++h) orow[16 * h + m] = (__bf16)acc[h][r];
  }
}

// ---------------------------------------------------------------------------
// Kernel 2: fused flash attention.  grid = BT/64 blocks, 128 threads (4 waves).
// Each wave owns 16 query rows; block streams 32 keys/values per iteration,
// double-buffered in LDS with async copies overlapping the WMMA compute.
// ---------------------------------------------------------------------------
__global__ __launch_bounds__(128)
void attn_kernel(const __bf16* __restrict__ qbuf,
                 const __bf16* __restrict__ kbuf,
                 const __bf16* __restrict__ vbuf,
                 float* __restrict__ out) {
  __shared__ __attribute__((aligned(16))) __bf16 kT[2][32][HDIM];   // [buf][key][h]
  __shared__ __attribute__((aligned(16))) __bf16 vT[2][HDIM][32];   // [buf][h][key]
  __shared__ __attribute__((aligned(16))) __bf16 pT[4][16][32];     // per-wave P

  const int tid  = threadIdx.x;
  const int wv   = tid >> 5;
  const int lane = tid & 31;
  const int m    = lane & 15;
  const int g    = lane >> 4;

  const int blockRow0 = blockIdx.x * 64;
  const int b         = blockRow0 / TLEN;       // whole block is in one batch
  const int rowBase   = blockRow0 + wv * 16;
  const size_t kvBase = (size_t)b * TLEN * HDIM;

  // per-thread staging coordinates: 32 keys x 64 h per tile
  const int skey = tid >> 2;            // 0..31
  const int shb  = (tid & 3) * 16;      // 0,16,32,48

  // stage K (async pure copy) + V (register transpose) for tile j into buf
  auto stage = [&](int j, int buf) {
    const __bf16* ks = kbuf + kvBase + (size_t)(j * 32 + skey) * HDIM + shb;
    async_copy_16B(ks,     &kT[buf][skey][shb]);
    async_copy_16B(ks + 8, &kT[buf][skey][shb + 8]);
    const __bf16* vs = vbuf + kvBase + (size_t)(j * 32 + skey) * HDIM + shb;
    v8bf v0 = *(const v8bf*)(vs);
    v8bf v1 = *(const v8bf*)(vs + 8);
#pragma unroll
    for (int i = 0; i < 8; ++i) {
      vT[buf][shb + i][skey]     = v0[i];
      vT[buf][shb + 8 + i][skey] = v1[i];
    }
  };

  // Q fragments: 16 rows x 64, two K-chunks of 32
  const __bf16* qrow = qbuf + (size_t)(rowBase + m) * HDIM;
  v16bf qa[2];
#pragma unroll
  for (int c = 0; c < 2; ++c)
    qa[c] = cat16(*(const v8bf*)(qrow + 32 * c + 8 * g),
                  *(const v8bf*)(qrow + 32 * c + 16 + 8 * g));

  float mrow[8], lrow[8];
  v8f acc[4] = {{}, {}, {}, {}};
#pragma unroll
  for (int r = 0; r < 8; ++r) { mrow[r] = -3.0e38f; lrow[r] = 0.0f; }

  constexpr int NJ = TLEN / 32;
  stage(0, 0);
  wait_async0();
  __syncthreads();

  for (int j = 0; j < NJ; ++j) {
    const int cur = j & 1;

    // ---- kick off next tile's staging into the other buffer ----
    if (j + 1 < NJ) {
      stage(j + 1, cur ^ 1);
      if (j + 2 < NJ) {               // gfx1250 global_prefetch_b8
        __builtin_prefetch(kbuf + kvBase + (size_t)((j + 2) * 32 + skey) * HDIM + shb, 0, 3);
        __builtin_prefetch(vbuf + kvBase + (size_t)((j + 2) * 32 + skey) * HDIM + shb, 0, 3);
      }
    }

    // ---- scores: S = q @ k^T for two 16-key sub-tiles ----
    v8f s0 = {}, s1 = {};
#pragma unroll
    for (int c = 0; c < 2; ++c) {
      v16bf kb0 = cat16(*(const v8bf*)&kT[cur][m][32 * c + 8 * g],
                        *(const v8bf*)&kT[cur][m][32 * c + 16 + 8 * g]);
      s0 = wmma_bf16(qa[c], kb0, s0);
      v16bf kb1 = cat16(*(const v8bf*)&kT[cur][16 + m][32 * c + 8 * g],
                        *(const v8bf*)&kT[cur][16 + m][32 * c + 16 + 8 * g]);
      s1 = wmma_bf16(qa[c], kb1, s1);
    }

    // ---- online softmax (row = 16 lanes of one half-wave) ----
#pragma unroll
    for (int r = 0; r < 8; ++r) {
      float a0 = s0[r] * SM_SCALE;
      float a1 = s1[r] * SM_SCALE;
      float mx = fmaxf(a0, a1);
#pragma unroll
      for (int off = 1; off < 16; off <<= 1)
        mx = fmaxf(mx, __shfl_xor(mx, off, 32));
      const float mnew = fmaxf(mrow[r], mx);
      const float p0 = __expf(a0 - mnew);
      const float p1 = __expf(a1 - mnew);
      float ps = p0 + p1;
#pragma unroll
      for (int off = 1; off < 16; off <<= 1)
        ps += __shfl_xor(ps, off, 32);
      const float alpha = __expf(mrow[r] - mnew);
      lrow[r] = lrow[r] * alpha + ps;
      mrow[r] = mnew;
#pragma unroll
      for (int h = 0; h < 4; ++h) acc[h][r] *= alpha;
      const int M = r + 8 * g;
      pT[wv][M][m]      = (__bf16)p0;
      pT[wv][M][16 + m] = (__bf16)p1;
    }
    // per-wave LDS round-trip: order ds writes before fragment reads
    asm volatile("s_wait_dscnt 0" ::: "memory");

    // ---- O += P @ V ----
    v16bf pa = cat16(*(const v8bf*)&pT[wv][m][8 * g],
                     *(const v8bf*)&pT[wv][m][16 + 8 * g]);
#pragma unroll
    for (int h = 0; h < 4; ++h) {
      v16bf vb = cat16(*(const v8bf*)&vT[cur][16 * h + m][8 * g],
                       *(const v8bf*)&vT[cur][16 * h + m][16 + 8 * g]);
      acc[h] = wmma_bf16(pa, vb, acc[h]);
    }

    // publish buffer j+1, retire reads of buffer j
    wait_async0();
    __syncthreads();
  }

  // ---- normalize and write fp32 output ----
#pragma unroll
  for (int r = 0; r < 8; ++r) {
    const float inv = 1.0f / lrow[r];
    float* orow = out + (size_t)(rowBase + r + 8 * g) * HDIM;
#pragma unroll
    for (int h = 0; h < 4; ++h) orow[16 * h + m] = acc[h][r] * inv;
  }
}

// ---------------------------------------------------------------------------
extern "C" void kernel_launch(void* const* d_in, const int* in_sizes, int n_in,
                              void* d_out, int out_size, void* d_ws, size_t ws_size,
                              hipStream_t stream) {
  const float* x  = (const float*)d_in[0];
  const float* Wk = (const float*)d_in[1];
  const float* Wq = (const float*)d_in[2];
  const float* Wv = (const float*)d_in[3];

  __bf16* kbuf = (__bf16*)d_ws;                 // [BT, H] bf16   (2 MB)
  __bf16* qbuf = kbuf + (size_t)BT * HDIM;      // [BT, H] bf16   (2 MB)
  __bf16* vbuf = qbuf + (size_t)BT * HDIM;      // [BT, H] bf16   (2 MB)

  dim3 pgrid(BT / 64, 3);
  proj_kernel<<<pgrid, 128, 0, stream>>>(x, Wk, Wq, Wv, kbuf, qbuf, vbuf);

  attn_kernel<<<BT / 64, 128, 0, stream>>>(qbuf, kbuf, vbuf, (float*)d_out);
}
